// SpGraphAttentionLayer_61615600828799
// MI455X (gfx1250) — matrix operand order repaired and checked by reading, not compiled, not run on hardware
//
#include <hip/hip_runtime.h>

typedef __attribute__((ext_vector_type(16))) __bf16 v16bf;
typedef __attribute__((ext_vector_type(8)))  float  v8f;

#define F_DIM 256
#define WPAD 16
#define WSTRIDE (F_DIM + WPAD)   // 272 bf16 per column -> 544B stride, 32B aligned
#define ALPHA 0.2f

// -------------------------------------------------------------------------
// Zero init: h' accumulator (d_out) and rowsum
// -------------------------------------------------------------------------
__global__ void zero2_kernel(float* __restrict__ a, long long na,
                             float* __restrict__ b, long long nb) {
  long long i = (long long)blockIdx.x * blockDim.x + threadIdx.x;
  if (i < na) a[i] = 0.0f;
  if (i < nb) b[i] = 0.0f;
}

// -------------------------------------------------------------------------
// h = x @ W via v_wmma_f32_16x16x32_bf16.
// Block = 256 threads = 8 waves; each wave computes a 16-row strip of h.
// W (fp32 row-major [K=256][N=256]) converted once to bf16, stored TRANSPOSED
// in LDS (padded 544B column stride) so B-fragments are contiguous 32B loads.
// Two column tiles per iteration -> two independent WMMA accumulator chains
// (fills the matrix pipe; overlaps DS latency of one chain with the other).
// -------------------------------------------------------------------------
__global__ __launch_bounds__(256) void gat_gemm_kernel(
    const float* __restrict__ x, const float* __restrict__ W,
    float* __restrict__ h, int n) {
  __shared__ __align__(32) __bf16 wT[F_DIM * WSTRIDE];  // 136 KB of 320 KB WGP LDS

  const int tid = threadIdx.x;

  // Cooperative load+convert+transpose of W: wT[ncol][k] = bf16(W[k][ncol])
  for (int i = tid; i < F_DIM * F_DIM; i += 256) {
    int k = i >> 8;
    int ncol = i & 255;
    wT[ncol * WSTRIDE + k] = (__bf16)W[i];
  }
  __syncthreads();

  const int wave = tid >> 5;
  const int lane = tid & 31;
  const int grp  = lane >> 4;   // lane group 0/1 per ISA A/B layout
  const int l16  = lane & 15;

  const int rowBase = blockIdx.x * 128 + wave * 16;
  const bool fullTile = (rowBase + 16 <= n);  // false only in the tail block
  int row = rowBase + l16;
  int rowClamped = (row < n) ? row : 0;       // keep EXEC all-ones for WMMA
  const float* xr = x + (size_t)rowClamped * F_DIM;

  // A fragments, all 8 K-steps resident (64 VGPRs).
  // A 16x32 (contiguous, matches B): grp0 holds K 0..15, grp1 holds K 16..31.
  v16bf afrag[8];
#pragma unroll
  for (int kk = 0; kk < 8; ++kk) {
    const float* p0 = xr + kk * 32 + grp * 16;
    const float* p1 = xr + kk * 32 + grp * 16 + 8;
#pragma unroll
    for (int j = 0; j < 8; ++j) {
      afrag[kk][j]     = (__bf16)p0[j];
      afrag[kk][8 + j] = (__bf16)p1[j];
    }
  }

  // 8 iterations x 2 column tiles x 8 K-steps = 128 WMMAs per wave.
  for (int nt = 0; nt < 16; nt += 2) {
    // B 32x16 bf16: grp0 lanes hold K 0..15, grp1 lanes hold K 16..31 (contig)
    const __bf16* bp0 = wT + (size_t)(nt * 16 + l16) * WSTRIDE + grp * 16;
    const __bf16* bp1 = bp0 + (size_t)16 * WSTRIDE;

    v8f c0 = {};
    v8f c1 = {};
#pragma unroll
    for (int kk = 0; kk < 8; ++kk) {
      v16bf b0 = *(const v16bf*)(bp0 + kk * 32);
      v16bf b1 = *(const v16bf*)(bp1 + kk * 32);
      c0 = __builtin_amdgcn_wmma_f32_16x16x32_bf16(
          false, afrag[kk], false, b0, (short)0, c0, false, false);
      c1 = __builtin_amdgcn_wmma_f32_16x16x32_bf16(
          false, afrag[kk], false, b1, (short)0, c1, false, false);
    }

    // C/D layout: VGPR r -> M = r + 8*grp, N = l16
    float* hp0 = h + (size_t)(rowBase + 8 * grp) * F_DIM + nt * 16 + l16;
    float* hp1 = hp0 + 16;
    if (fullTile) {
#pragma unroll
      for (int r = 0; r < 8; ++r) {
        hp0[(size_t)r * F_DIM] = c0[r];
        hp1[(size_t)r * F_DIM] = c1[r];
      }
    } else {
#pragma unroll
      for (int r = 0; r < 8; ++r) {
        int m = rowBase + r + 8 * grp;
        if (m < n) {
          hp0[(size_t)r * F_DIM] = c0[r];
          hp1[(size_t)r * F_DIM] = c1[r];
        }
      }
    }
  }
}

// -------------------------------------------------------------------------
// s = h @ a[:256], t = h @ a[256:]  — one wave per node row
// -------------------------------------------------------------------------
__global__ __launch_bounds__(256) void gat_st_kernel(
    const float* __restrict__ h, const float* __restrict__ a,
    float* __restrict__ s, float* __restrict__ t, int n) {
  int node = (blockIdx.x * 256 + threadIdx.x) >> 5;
  int lane = threadIdx.x & 31;
  if (node >= n) return;
  const float* hr = h + (size_t)node * F_DIM;
  float ss = 0.0f, tt = 0.0f;
#pragma unroll
  for (int j = lane; j < F_DIM; j += 32) {
    float hv = hr[j];
    ss += hv * a[j];
    tt += hv * a[F_DIM + j];
  }
#pragma unroll
  for (int off = 16; off > 0; off >>= 1) {
    ss += __shfl_xor(ss, off, 32);
    tt += __shfl_xor(tt, off, 32);
  }
  if (lane == 0) { s[node] = ss; t[node] = tt; }
}

// -------------------------------------------------------------------------
// Edge scatter: one wave per edge. e = exp(-leakyrelu(s[src]+t[dst]));
// rowsum[src] += e; h'[src] += e * h[dst]. h (51 MB) lives in the 192 MB L2,
// so gathers + no-return f32 atomics resolve in L2, not HBM.
// -------------------------------------------------------------------------
__global__ __launch_bounds__(256) void gat_edge_kernel(
    const int* __restrict__ src, const int* __restrict__ dst,
    const float* __restrict__ s, const float* __restrict__ t,
    const float* __restrict__ h, float* __restrict__ acc,
    float* __restrict__ rowsum, int E) {
  int e = (blockIdx.x * 256 + threadIdx.x) >> 5;
  int lane = threadIdx.x & 31;
  if (e >= E) return;

  int u = src[e];
  int v = dst[e];
  const float* hv = h + (size_t)v * F_DIM;
  __builtin_prefetch(hv + lane * 8, 0, 0);   // global_prefetch_b8 ahead of gather

  float logit = s[u] + t[v];
  float lr = (logit > 0.0f) ? logit : ALPHA * logit;
  float w = __expf(-lr);
  if (lane == 0) atomicAdd(rowsum + u, w);

  float* out = acc + (size_t)u * F_DIM;
  float4 a0 = ((const float4*)hv)[lane];        // covers floats [0,128)
  float4 a1 = ((const float4*)hv)[lane + 32];   // covers floats [128,256)
  int b = lane * 4;
  atomicAdd(out + b + 0,   w * a0.x);
  atomicAdd(out + b + 1,   w * a0.y);
  atomicAdd(out + b + 2,   w * a0.z);
  atomicAdd(out + b + 3,   w * a0.w);
  atomicAdd(out + b + 128, w * a1.x);
  atomicAdd(out + b + 129, w * a1.y);
  atomicAdd(out + b + 130, w * a1.z);
  atomicAdd(out + b + 131, w * a1.w);
}

// -------------------------------------------------------------------------
// h' = relu(h' / rowsum)  in place on d_out
// -------------------------------------------------------------------------
__global__ void gat_norm_kernel(float* __restrict__ out,
                                const float* __restrict__ rowsum,
                                long long ntot) {
  long long i = (long long)blockIdx.x * blockDim.x + threadIdx.x;
  if (i >= ntot) return;
  int row = (int)(i >> 8);
  float v = out[i] / rowsum[row];
  out[i] = v > 0.0f ? v : 0.0f;
}

// -------------------------------------------------------------------------
extern "C" void kernel_launch(void* const* d_in, const int* in_sizes, int n_in,
                              void* d_out, int out_size, void* d_ws, size_t ws_size,
                              hipStream_t stream) {
  const float* x    = (const float*)d_in[0];
  const int*   edge = (const int*)d_in[1];
  const float* W    = (const float*)d_in[2];
  const float* a    = (const float*)d_in[3];

  const int n = in_sizes[0] / F_DIM;   // 50000
  const int E = in_sizes[1] / 2;       // 850000
  float* out = (float*)d_out;

  // workspace layout: h [n*256] | s [n] | t [n] | rowsum [n]
  float* h      = (float*)d_ws;
  float* s      = h + (size_t)n * F_DIM;
  float* t      = s + n;
  float* rowsum = t + n;

  const long long ntot = (long long)n * F_DIM;

  zero2_kernel<<<(int)((ntot + 255) / 256), 256, 0, stream>>>(out, ntot, rowsum, n);
  gat_gemm_kernel<<<(n + 127) / 128, 256, 0, stream>>>(x, W, h, n);
  gat_st_kernel<<<(n + 7) / 8, 256, 0, stream>>>(h, a, s, t, n);
  gat_edge_kernel<<<(E + 7) / 8, 256, 0, stream>>>(edge, edge + E, s, t, h, out,
                                                   rowsum, E);
  gat_norm_kernel<<<(int)((ntot + 255) / 256), 256, 0, stream>>>(out, rowsum, ntot);
}
